// MixtralAttention_37709812859144
// MI455X (gfx1250) — matrix operand order repaired and checked by reading, not compiled, not run on hardware
//
#include <hip/hip_runtime.h>
#include <hip/hip_bf16.h>
#include <stdint.h>

#define SEQ    2048
#define HIDDEN 4096
#define NHEADS 32
#define NKV    8
#define HD     128

typedef __bf16 bf16x16 __attribute__((ext_vector_type(16)));
typedef float  f32x8   __attribute__((ext_vector_type(8)));
typedef float  f32x4   __attribute__((ext_vector_type(4)));
typedef unsigned short u16x8 __attribute__((ext_vector_type(8)));
typedef unsigned short u16x4 __attribute__((ext_vector_type(4)));

union Frag { bf16x16 v; uint32_t w[8]; unsigned short u[16]; };

__device__ __forceinline__ unsigned short f2bf(float f) {
  union { float f; uint32_t u; } c; c.f = f;
  uint32_t u = c.u;
  uint32_t rb = 0x7FFFu + ((u >> 16) & 1u);     // round-to-nearest-even
  return (unsigned short)((u + rb) >> 16);
}

// u32-granule offset for the 16-bit A-matrix layout:
// lane group 0 holds K {0..7, 16..23}, lane group 1 holds +8.
__device__ __forceinline__ int aOff(int v, int lh) {
  return v + ((v >= 4) ? 4 : 0) + 4 * lh;
}

// Low 32 bits of a flat shared pointer == LDS byte address (aperture rule).
__device__ __forceinline__ uint32_t lds_addr(const void* p) {
  return (uint32_t)(uintptr_t)p;
}

// Async bf16 tile copy: 16B granules, ASYNCcnt-tracked, no VGPR round trip.
__device__ __forceinline__ void async_copy_b128_x2(uint32_t lds, uint64_t ga) {
  asm volatile("global_load_async_to_lds_b128 %0, %1, off\n\t"
               "global_load_async_to_lds_b128 %0, %1, off offset:16"
               :: "v"(lds), "v"(ga) : "memory");
}
__device__ __forceinline__ void async_copy_b128_x4(uint32_t lds, uint64_t ga) {
  asm volatile("global_load_async_to_lds_b128 %0, %1, off\n\t"
               "global_load_async_to_lds_b128 %0, %1, off offset:16\n\t"
               "global_load_async_to_lds_b128 %0, %1, off offset:32\n\t"
               "global_load_async_to_lds_b128 %0, %1, off offset:48"
               :: "v"(lds), "v"(ga) : "memory");
}
__device__ __forceinline__ void wait_async0() {
  asm volatile("s_wait_asynccnt 0" ::: "memory");
}

// ---------------------------------------------------------------------------
// GEMM: C[M,N] = A[M,K] (f32 or bf16) * B[K,N] (f32), compute in bf16 WMMA,
// f32 accumulate. Block tile 128x64, BK=32, 8 waves of 32x32 each.
// bf16 A tiles are staged with async global->LDS DMA.
// ---------------------------------------------------------------------------
template <typename AT, bool OUTBF>
__global__ __launch_bounds__(256) void gemm_wmma(const AT* __restrict__ A,
                                                 const float* __restrict__ B,
                                                 void* __restrict__ Cv,
                                                 int M, int N, int K) {
  __shared__ unsigned short As[128 * 40];   // [row][k] stride 40 (pad vs banks)
  __shared__ unsigned short Bt[64 * 40];    // transposed: [n][k] stride 40

  const int tid  = threadIdx.x;
  const int wave = tid >> 5, lane = tid & 31;
  const int wm = wave >> 1, wn = wave & 1;
  const int lh = lane >> 4, ln = lane & 15;
  const int mb = blockIdx.y, nb = blockIdx.x;

  f32x8 acc[2][2] = {};

  const int arow = tid >> 1, acb = (tid & 1) * 16;   // A staging: 16 elems/thread
  const int brow = tid >> 3, bnb = (tid & 7) * 8;    // B staging: 8 elems/thread

  const uint32_t* As32 = (const uint32_t*)As;
  const uint32_t* Bt32 = (const uint32_t*)Bt;
  const uint32_t aLds = lds_addr(&As[arow * 40 + acb]);

  for (int k0 = 0; k0 < K; k0 += 32) {
    // ---- stage A tile ----
    if constexpr (sizeof(AT) == 4) {
      const f32x4* src = (const f32x4*)(A + (size_t)(mb * 128 + arow) * K + k0 + acb);
#pragma unroll
      for (int i = 0; i < 4; ++i) {
        f32x4 x = src[i];
        u16x4 y = { f2bf(x[0]), f2bf(x[1]), f2bf(x[2]), f2bf(x[3]) };
        *(u16x4*)&As[arow * 40 + acb + 4 * i] = y;
      }
    } else {
      // bf16 in memory: async DMA straight into LDS (2 x B128 per thread).
      async_copy_b128_x2(aLds,
          (uint64_t)(uintptr_t)(A + (size_t)(mb * 128 + arow) * K + k0 + acb));
    }
    // ---- stage B tile transposed (f32 -> bf16 conversion in registers) ----
    {
      const f32x4* src = (const f32x4*)(B + (size_t)(k0 + brow) * N + nb * 64 + bnb);
      f32x4 x0 = src[0], x1 = src[1];
#pragma unroll
      for (int e = 0; e < 4; ++e) Bt[(bnb + e) * 40 + brow] = f2bf(x0[e]);
#pragma unroll
      for (int e = 0; e < 4; ++e) Bt[(bnb + 4 + e) * 40 + brow] = f2bf(x1[e]);
    }
    // prefetch next K-step tiles into cache while this tile computes
    if (k0 + 32 < K) {
      __builtin_prefetch(A + (size_t)(mb * 128 + arow) * K + (k0 + 32) + acb, 0, 1);
      __builtin_prefetch(B + (size_t)(k0 + 32 + brow) * N + nb * 64 + bnb, 0, 1);
    }
    if constexpr (sizeof(AT) != 4) wait_async0();
    __syncthreads();

    Frag a[2], b[2];
#pragma unroll
    for (int i = 0; i < 2; ++i) {
      int row = wm * 32 + i * 16 + ln;
#pragma unroll
      for (int v = 0; v < 8; ++v) a[i].w[v] = As32[row * 20 + aOff(v, lh)];
    }
#pragma unroll
    for (int j = 0; j < 2; ++j) {
      int col = wn * 32 + j * 16 + ln;
#pragma unroll
      for (int v = 0; v < 8; ++v) b[j].w[v] = Bt32[col * 20 + aOff(v, lh)];
    }
#pragma unroll
    for (int i = 0; i < 2; ++i)
#pragma unroll
      for (int j = 0; j < 2; ++j)
        acc[i][j] = __builtin_amdgcn_wmma_f32_16x16x32_bf16(
            false, a[i].v, false, b[j].v, (short)0, acc[i][j], false, false);
    __syncthreads();
  }

  // ---- epilogue: C layout VGPR r -> row r + 8*lh, col = ln ----
#pragma unroll
  for (int i = 0; i < 2; ++i)
#pragma unroll
    for (int j = 0; j < 2; ++j)
#pragma unroll
      for (int r = 0; r < 8; ++r) {
        int row = mb * 128 + wm * 32 + i * 16 + r + 8 * lh;
        int col = nb * 64 + wn * 32 + j * 16 + ln;
        if constexpr (OUTBF)
          ((unsigned short*)Cv)[(size_t)row * N + col] = f2bf(acc[i][j][r]);
        else
          ((float*)Cv)[(size_t)row * N + col] = acc[i][j][r];
      }
}

// ---------------------------------------------------------------------------
// RoPE: read f32 projections, write bf16 rotated values.
// ---------------------------------------------------------------------------
__global__ void rope_kernel(const int* __restrict__ pos, const float* __restrict__ X,
                            unsigned short* __restrict__ O, int nheads) {
  int idx = blockIdx.x * blockDim.x + threadIdx.x;
  int total = SEQ * nheads * 64;
  if (idx >= total) return;
  int d = idx & 63;
  int h = (idx >> 6) % nheads;
  int t = idx / (64 * nheads);
  float p = (float)pos[t];
  // inv_freq = ROPE_BASE^(-d/64), ln(1e6)/64 = 0.2158673524681918
  float inv = __expf(-(float)d * 0.2158673524681918f);
  float ang = p * inv;
  float c = cosf(ang), s = sinf(ang);
  size_t base = (size_t)t * (nheads * 128) + (size_t)h * 128 + d;
  float x1 = X[base], x2 = X[base + 64];
  O[base]      = f2bf(x1 * c - x2 * s);
  O[base + 64] = f2bf(x2 * c + x1 * s);
}

// ---------------------------------------------------------------------------
// Flash attention: 4 waves/WG, each wave owns one 16-row Q tile of one head.
// K chunks staged by async global->LDS DMA; V transposed during staging;
// online softmax; all matrix math in bf16 WMMA with f32 accumulation.
// ---------------------------------------------------------------------------
__global__ __launch_bounds__(128) void attn_kernel(const unsigned short* __restrict__ Qb,
                                                   const unsigned short* __restrict__ Kb,
                                                   const unsigned short* __restrict__ Vb,
                                                   unsigned short* __restrict__ AOb) {
  __shared__ unsigned short Ks[32 * 136];   // [key][d]  stride 136
  __shared__ unsigned short Vt[128 * 40];   // [d][key]  stride 40 (transposed)
  __shared__ unsigned short Ps[4 * 16 * 40];// per-wave P tile [row][key]

  const int tid  = threadIdx.x;
  const int wave = tid >> 5, lane = tid & 31;
  const int lh = lane >> 4, ln = lane & 15;
  const int h = blockIdx.y, g = h >> 2;
  const int qb = blockIdx.x;
  const int qbase = qb * 64 + wave * 16;

  const uint32_t* Ks32 = (const uint32_t*)Ks;
  const uint32_t* Vt32 = (const uint32_t*)Vt;
  const uint32_t* Ps32 = (const uint32_t*)Ps;

  // Q A-fragments straight from global (row-major matches A layout).
  Frag qf[4];
  {
    const uint32_t* qp =
        (const uint32_t*)(Qb + (size_t)(qbase + ln) * (NHEADS * HD) + (size_t)h * HD);
#pragma unroll
    for (int kk = 0; kk < 4; ++kk)
#pragma unroll
      for (int v = 0; v < 8; ++v) qf[kk].w[v] = qp[16 * kk + aOff(v, lh)];
  }

  f32x8 accO[8] = {};
  float mrow[8], srow[8];
#pragma unroll
  for (int r = 0; r < 8; ++r) { mrow[r] = -1e30f; srow[r] = 0.f; }

  const float scale = 0.08838834764831845f;  // 1/sqrt(128)
  const int nch = qb * 2 + 2;
  const int kvs = NKV * HD;

  const int key = tid >> 2;
  const int db  = (tid & 3) * 32;
  const uint32_t kLds = lds_addr(&Ks[key * 136 + db]);

  for (int c = 0; c < nch; ++c) {
    const int kstart = c * 32;
    // ---- K chunk: async DMA global -> LDS (4 x B128 per thread) ----
    async_copy_b128_x4(kLds,
        (uint64_t)(uintptr_t)(Kb + (size_t)(kstart + key) * kvs + g * HD + db));
    // ---- V chunk: load + transpose into LDS ----
    {
      const u16x8* vsrc = (const u16x8*)(Vb + (size_t)(kstart + key) * kvs + g * HD + db);
#pragma unroll
      for (int i = 0; i < 4; ++i) {
        u16x8 vv = vsrc[i];
#pragma unroll
        for (int e = 0; e < 8; ++e) Vt[(db + 8 * i + e) * 40 + key] = vv[e];
      }
    }
    // prefetch next chunk while this one computes
    if (c + 1 < nch) {
      __builtin_prefetch(Kb + (size_t)(kstart + 32 + key) * kvs + g * HD + db, 0, 1);
      __builtin_prefetch(Vb + (size_t)(kstart + 32 + key) * kvs + g * HD + db, 0, 1);
    }
    wait_async0();
    __syncthreads();

    if (kstart <= qbase + 15) {          // wave-uniform: EXEC stays all-ones
      // ---- S = Q * K^T over the 32-key chunk (2 tiles of 16 keys) ----
      f32x8 s[2] = {};
#pragma unroll
      for (int j = 0; j < 2; ++j)
#pragma unroll
        for (int kk = 0; kk < 4; ++kk) {
          Frag kb;
#pragma unroll
          for (int v = 0; v < 8; ++v)
            kb.w[v] = Ks32[(16 * j + ln) * 68 + 16 * kk + aOff(v, lh)];
          s[j] = __builtin_amdgcn_wmma_f32_16x16x32_bf16(
              false, qf[kk].v, false, kb.v, (short)0, s[j], false, false);
        }

      // ---- scale + causal mask + online softmax ----
      float p[2][8], mx[8];
#pragma unroll
      for (int r = 0; r < 8; ++r) {
        int row = qbase + r + 8 * lh;
        float v0 = s[0][r] * scale; if (kstart + ln > row)      v0 = -1e30f;
        float v1 = s[1][r] * scale; if (kstart + 16 + ln > row) v1 = -1e30f;
        p[0][r] = v0; p[1][r] = v1;
        mx[r] = fmaxf(v0, v1);
      }
#pragma unroll
      for (int r = 0; r < 8; ++r)
#pragma unroll
        for (int off = 1; off <= 8; off <<= 1)
          mx[r] = fmaxf(mx[r], __shfl_xor(mx[r], off, 32));

      float alpha[8];
#pragma unroll
      for (int r = 0; r < 8; ++r) {
        float mnew = fmaxf(mrow[r], mx[r]);
        alpha[r] = __expf(mrow[r] - mnew);
        mrow[r] = mnew;
        float e0 = __expf(p[0][r] - mnew);
        float e1 = __expf(p[1][r] - mnew);
        p[0][r] = e0; p[1][r] = e1;
        float rs = e0 + e1;
#pragma unroll
        for (int off = 1; off <= 8; off <<= 1) rs += __shfl_xor(rs, off, 32);
        srow[r] = srow[r] * alpha[r] + rs;
      }
#pragma unroll
      for (int t = 0; t < 8; ++t)
#pragma unroll
        for (int r = 0; r < 8; ++r) accO[t][r] *= alpha[r];

      // ---- P through wave-private LDS to reshape C-layout -> A-layout ----
#pragma unroll
      for (int j = 0; j < 2; ++j)
#pragma unroll
        for (int r = 0; r < 8; ++r)
          Ps[wave * 640 + (r + 8 * lh) * 40 + 16 * j + ln] = f2bf(p[j][r]);
      asm volatile("s_wait_dscnt 0" ::: "memory");  // DS in-order per wave

      Frag pf;
#pragma unroll
      for (int v = 0; v < 8; ++v)
        pf.w[v] = Ps32[wave * 320 + ln * 20 + aOff(v, lh)];

      // ---- O += P * V (8 tiles over the 128-wide head dim) ----
#pragma unroll
      for (int t = 0; t < 8; ++t) {
        Frag vb;
#pragma unroll
        for (int v = 0; v < 8; ++v)
          vb.w[v] = Vt32[(16 * t + ln) * 20 + aOff(v, lh)];
        accO[t] = __builtin_amdgcn_wmma_f32_16x16x32_bf16(
            false, pf.v, false, vb.v, (short)0, accO[t], false, false);
      }
    }
    __syncthreads();  // unconditional: all waves see same barrier count
  }

  // ---- normalize and store bf16 attention output ----
  float inv[8];
#pragma unroll
  for (int r = 0; r < 8; ++r) inv[r] = 1.0f / srow[r];
#pragma unroll
  for (int t = 0; t < 8; ++t)
#pragma unroll
    for (int r = 0; r < 8; ++r) {
      int row = qbase + r + 8 * lh;
      int col = h * HD + 16 * t + ln;
      AOb[(size_t)row * (NHEADS * HD) + col] = f2bf(accO[t][r] * inv[r]);
    }
}

// ---------------------------------------------------------------------------
extern "C" void kernel_launch(void* const* d_in, const int* in_sizes, int n_in,
                              void* d_out, int out_size, void* d_ws, size_t ws_size,
                              hipStream_t stream) {
  (void)in_sizes; (void)n_in; (void)out_size; (void)ws_size;
  const int*   positions = (const int*)d_in[0];
  const float* X  = (const float*)d_in[1];
  const float* Wq = (const float*)d_in[2];
  const float* Wk = (const float*)d_in[3];
  const float* Wv = (const float*)d_in[4];
  const float* Wo = (const float*)d_in[5];
  float* out = (float*)d_out;

  char* ws = (char*)d_ws;
  float* Qf = (float*)ws;                    ws += (size_t)SEQ * HIDDEN * 4;   // 32 MiB
  float* Kf = (float*)ws;                    ws += (size_t)SEQ * NKV * HD * 4; //  8 MiB
  unsigned short* Qb = (unsigned short*)ws;  ws += (size_t)SEQ * HIDDEN * 2;   // 16 MiB
  unsigned short* Kb = (unsigned short*)ws;  ws += (size_t)SEQ * NKV * HD * 2; //  4 MiB
  unsigned short* Vb = (unsigned short*)ws;  ws += (size_t)SEQ * NKV * HD * 2; //  4 MiB
  unsigned short* AO = (unsigned short*)ws;  ws += (size_t)SEQ * HIDDEN * 2;   // 16 MiB

  dim3 blk(256);
  gemm_wmma<float, false><<<dim3(HIDDEN / 64, SEQ / 128), blk, 0, stream>>>(
      X, Wq, Qf, SEQ, HIDDEN, HIDDEN);
  gemm_wmma<float, false><<<dim3((NKV * HD) / 64, SEQ / 128), blk, 0, stream>>>(
      X, Wk, Kf, SEQ, NKV * HD, HIDDEN);
  gemm_wmma<float, true><<<dim3((NKV * HD) / 64, SEQ / 128), blk, 0, stream>>>(
      X, Wv, Vb, SEQ, NKV * HD, HIDDEN);

  rope_kernel<<<(SEQ * NHEADS * 64) / 256, 256, 0, stream>>>(positions, Qf, Qb, NHEADS);
  rope_kernel<<<(SEQ * NKV * 64) / 256, 256, 0, stream>>>(positions, Kf, Kb, NKV);

  attn_kernel<<<dim3(SEQ / 64, NHEADS), dim3(128), 0, stream>>>(Qb, Kb, Vb, AO);

  gemm_wmma<unsigned short, false><<<dim3(HIDDEN / 64, SEQ / 128), blk, 0, stream>>>(
      AO, Wo, out, SEQ, HIDDEN, HIDDEN);
}
